// Self_Attention_Interview_18425409700433
// MI455X (gfx1250) — matrix operand order repaired and checked by reading, not compiled, not run on hardware
//
#include <hip/hip_runtime.h>
#include <hip/hip_bf16.h>

#define BATCH 4
#define SEQ   2048
#define DIM   1024

typedef __bf16 bf16_t;
typedef __attribute__((ext_vector_type(16))) __bf16 v16bf;
typedef __attribute__((ext_vector_type(8)))  __bf16 v8bf;
typedef __attribute__((ext_vector_type(8)))  float  v8f;
typedef __attribute__((ext_vector_type(4)))  float  v4f;

#if defined(__has_builtin)
#  if __has_builtin(__builtin_amdgcn_sched_barrier)
#    define SCHED_FENCE() __builtin_amdgcn_sched_barrier(0)
#  endif
#endif
#ifndef SCHED_FENCE
#  define SCHED_FENCE() do {} while (0)
#endif

union V16 {
    v16bf v;
    struct { v8bf lo, hi; } h;
};
static_assert(sizeof(v16bf) == 32, "v16bf must be 8 VGPRs");

__device__ __forceinline__ v8f wmma_bf16(const V16& a, const V16& b, v8f c) {
    return __builtin_amdgcn_wmma_f32_16x16x32_bf16(
        false, a.v, false, b.v, (short)0, c, false, false);
}

// One pipeline stage of fragments for a 32x64 output tile:
// two A fragments (M rows 0-15 / 16-31) + four B fragments (4x16 cols).
struct Stage {
    V16 a0, a1, b0, b1, b2, b3;
};

// Load a stage directly from bf16 row-major data.
// A frag: lane holds M=lane&15, K = half*16 + {0..15}.
// B frag: lane holds N=lane&15, K = half*16 + {0..15}.
__device__ __forceinline__ void load_stage(
    Stage& s,
    const bf16_t* q0, const bf16_t* q1,
    const bf16_t* k0, const bf16_t* k1,
    const bf16_t* k2, const bf16_t* k3,
    int off_a, int off_b)
{
    s.a0.h.lo = *(const v8bf*)(q0 + off_b);
    s.a0.h.hi = *(const v8bf*)(q0 + off_b + 8);
    s.a1.h.lo = *(const v8bf*)(q1 + off_b);
    s.a1.h.hi = *(const v8bf*)(q1 + off_b + 8);
    s.b0.h.lo = *(const v8bf*)(k0 + off_b);
    s.b0.h.hi = *(const v8bf*)(k0 + off_b + 8);
    s.b1.h.lo = *(const v8bf*)(k1 + off_b);
    s.b1.h.hi = *(const v8bf*)(k1 + off_b + 8);
    s.b2.h.lo = *(const v8bf*)(k2 + off_b);
    s.b2.h.hi = *(const v8bf*)(k2 + off_b + 8);
    s.b3.h.lo = *(const v8bf*)(k3 + off_b);
    s.b3.h.hi = *(const v8bf*)(k3 + off_b + 8);
}

__device__ __forceinline__ void mma_stage(const Stage& s, v8f (&acc)[2][4]) {
    acc[0][0] = wmma_bf16(s.a0, s.b0, acc[0][0]);
    acc[0][1] = wmma_bf16(s.a0, s.b1, acc[0][1]);
    acc[0][2] = wmma_bf16(s.a0, s.b2, acc[0][2]);
    acc[0][3] = wmma_bf16(s.a0, s.b3, acc[0][3]);
    acc[1][0] = wmma_bf16(s.a1, s.b0, acc[1][0]);
    acc[1][1] = wmma_bf16(s.a1, s.b1, acc[1][1]);
    acc[1][2] = wmma_bf16(s.a1, s.b2, acc[1][2]);
    acc[1][3] = wmma_bf16(s.a1, s.b3, acc[1][3]);
}

// ---------------------------------------------------------------------------
// Kernel 1: fused QKV projection (fp32 in, bf16 out).
// One wave computes a 16x64 tile.  A sched fence after the raw-load block
// forces all ~20 b128 loads into flight (distinct VGPRs) before any
// bf16 conversion executes -> one loadcnt wait per K-step, not per load.
// Q,K stored row-major bf16; V stored transposed per batch: Vt[b][d][s].
// ---------------------------------------------------------------------------
__global__ __launch_bounds__(256) void qkv_proj_kernel(
    const float* __restrict__ X,
    const float* __restrict__ Wq, const float* __restrict__ bq,
    const float* __restrict__ Wk, const float* __restrict__ bk,
    const float* __restrict__ Wv, const float* __restrict__ bv,
    bf16_t* __restrict__ qbuf, bf16_t* __restrict__ kbuf,
    bf16_t* __restrict__ vtbuf)
{
    const int lane = threadIdx.x & 31;
    const int wid  = blockIdx.x * 8 + (threadIdx.x >> 5); // 0 .. 24575
    const int mat  = wid / (512 * 16);                    // 0=Q 1=K 2=V
    const int rem  = wid % (512 * 16);
    const int rt   = rem / 16;                            // 16-row tile of B*S
    const int cg   = rem % 16;                            // 64-col group of D

    const float* W    = (mat == 0) ? Wq : (mat == 1) ? Wk : Wv;
    const float* bias = (mat == 0) ? bq : (mat == 1) ? bk : bv;

    const int    half = lane >> 4;
    const size_t arow = (size_t)(rt * 16 + (lane & 15)) * DIM;
    const float* wrow[4];
    #pragma unroll
    for (int t = 0; t < 4; t++) {
        const int col = cg * 64 + t * 16 + (lane & 15);
        wrow[t] = W + (size_t)col * DIM;
    }

    v8f acc[4] = {};
    for (int kt = 0; kt < DIM; kt += 32) {
        // ---- batched raw loads (kept together by the fence) ----
        const float* xp = X + arow + kt + half * 8;
        __builtin_prefetch(xp + 32, 0, 0);
        v4f ar[4];
        ar[0] = *(const v4f*)(xp);
        ar[1] = *(const v4f*)(xp + 4);
        ar[2] = *(const v4f*)(xp + 16);
        ar[3] = *(const v4f*)(xp + 20);
        v4f br[4][4];
        #pragma unroll
        for (int t = 0; t < 4; t++) {
            const float* wp = wrow[t] + kt + half * 8;
            #pragma unroll
            for (int c = 0; c < 4; c++) br[t][c] = *(const v4f*)(wp + (c & 1) * 4 + (c >> 1) * 16);
        }
        SCHED_FENCE();      // all loads issued before any conversion
        // ---- convert to bf16 fragments ----
        V16 a;
        #pragma unroll
        for (int c = 0; c < 4; c++)
            #pragma unroll
            for (int i = 0; i < 4; i++) a.v[c * 4 + i] = (bf16_t)ar[c][i];
        V16 bm[4];
        #pragma unroll
        for (int t = 0; t < 4; t++)
            #pragma unroll
            for (int c = 0; c < 4; c++)
                #pragma unroll
                for (int i = 0; i < 4; i++)
                    bm[t].v[c * 4 + i] = (bf16_t)br[t][c][i];
        // ---- 4 WMMAs ----
        #pragma unroll
        for (int t = 0; t < 4; t++) acc[t] = wmma_bf16(a, bm[t], acc[t]);
        SCHED_FENCE();
    }

    // C/D fragment: VGPR r -> M = r + half*8, N = lane&15
    const int rowbase = rt * 16 + half * 8;               // global b*S + s
    const int b       = rowbase >> 11;
    #pragma unroll
    for (int t = 0; t < 4; t++) {
        const int   col = cg * 64 + t * 16 + (lane & 15);
        const float bv_ = bias[col];
        #pragma unroll
        for (int r = 0; r < 8; r++) {
            const int   row = rowbase + r;
            const float val = acc[t][r] + bv_;
            if (mat == 0) {
                qbuf[(size_t)row * DIM + col] = (bf16_t)val;
            } else if (mat == 1) {
                kbuf[(size_t)row * DIM + col] = (bf16_t)val;
            } else {
                const int s = row & (SEQ - 1);
                vtbuf[((size_t)b * DIM + col) * SEQ + s] = (bf16_t)val;
            }
        }
    }
}

// ---------------------------------------------------------------------------
// Kernel 2: scores[b][q][k] = (Q.Kt)/sqrt(D).  One wave = 32x64 tile.
// Double-buffered stages with sched fences so the 12 fragment loads of
// step n+1 stay in flight across the 8 WMMAs of step n.
// ---------------------------------------------------------------------------
__global__ __launch_bounds__(256) void scores_kernel(
    const bf16_t* __restrict__ qbuf, const bf16_t* __restrict__ kbuf,
    float* __restrict__ sc)
{
    const int lane = threadIdx.x & 31;
    const int wid  = blockIdx.x * 8 + (threadIdx.x >> 5); // 0 .. 8191
    const int b    = wid / (64 * 32);
    const int rem  = wid % (64 * 32);
    const int qt   = rem / 32;                            // 32-row query tile
    const int cg   = rem % 32;                            // 64-key group
    const int half = lane >> 4;

    const bf16_t* q0 = qbuf + ((size_t)(b * SEQ + qt * 32 + (lane & 15))) * DIM;
    const bf16_t* q1 = q0 + (size_t)16 * DIM;
    const bf16_t* kr[4];
    #pragma unroll
    for (int t = 0; t < 4; t++) {
        const int kc = cg * 64 + t * 16 + (lane & 15);
        kr[t] = kbuf + ((size_t)(b * SEQ + kc)) * DIM;
    }

    v8f acc[2][4] = {};
    Stage s0, s1;
    load_stage(s0, q0, q1, kr[0], kr[1], kr[2], kr[3], half * 8, half * 16);
    SCHED_FENCE();
    for (int kt = 0; kt < DIM; kt += 64) {
        load_stage(s1, q0, q1, kr[0], kr[1], kr[2], kr[3],
                   kt + 32 + half * 8, kt + 32 + half * 16);
        SCHED_FENCE();          // s1 loads in flight before s0 WMMAs
        mma_stage(s0, acc);
        SCHED_FENCE();
        if (kt + 64 < DIM)
            load_stage(s0, q0, q1, kr[0], kr[1], kr[2], kr[3],
                       kt + 64 + half * 8, kt + 64 + half * 16);
        SCHED_FENCE();          // s0 loads in flight before s1 WMMAs
        mma_stage(s1, acc);
        SCHED_FENCE();
    }

    const float scale = 0.03125f;                         // 1/sqrt(1024)
    #pragma unroll
    for (int j = 0; j < 2; j++)
        #pragma unroll
        for (int t = 0; t < 4; t++) {
            const int kc = cg * 64 + t * 16 + (lane & 15);
            #pragma unroll
            for (int r = 0; r < 8; r++) {
                const int qr = qt * 32 + j * 16 + half * 8 + r;
                sc[((size_t)b * SEQ + qr) * SEQ + kc] = acc[j][t][r] * scale;
            }
        }
}

// ---------------------------------------------------------------------------
// Kernel 3: masked row softmax, one 256-thread block per (b,q) row.
// ---------------------------------------------------------------------------
__global__ __launch_bounds__(256) void softmax_kernel(
    const float* __restrict__ sc, const int* __restrict__ mask,
    bf16_t* __restrict__ p)
{
    __shared__ float red[8];
    const size_t row  = blockIdx.x;                       // b*S + q
    const float* srow = sc + row * SEQ;
    const int*   mrow = mask + row * SEQ;
    bf16_t*      prow = p + row * SEQ;
    const int tid = threadIdx.x;
    const int lane = tid & 31, wv = tid >> 5;

    float x[8];
    #pragma unroll
    for (int i = 0; i < 8; i++) {
        const int j = tid + i * 256;
        const float v = srow[j];
        x[i] = (mrow[j] == 0) ? -__builtin_inff() : v;
    }

    float lm = x[0];
    #pragma unroll
    for (int i = 1; i < 8; i++) lm = fmaxf(lm, x[i]);
    #pragma unroll
    for (int o = 16; o > 0; o >>= 1) lm = fmaxf(lm, __shfl_xor(lm, o, 32));
    if (lane == 0) red[wv] = lm;
    __syncthreads();
    float rowmax = red[0];
    #pragma unroll
    for (int i = 1; i < 8; i++) rowmax = fmaxf(rowmax, red[i]);

    float e[8];
    float ls = 0.f;
    #pragma unroll
    for (int i = 0; i < 8; i++) { e[i] = __expf(x[i] - rowmax); ls += e[i]; }
    #pragma unroll
    for (int o = 16; o > 0; o >>= 1) ls += __shfl_xor(ls, o, 32);
    __syncthreads();
    if (lane == 0) red[wv] = ls;
    __syncthreads();
    float rowsum = 0.f;
    #pragma unroll
    for (int i = 0; i < 8; i++) rowsum += red[i];
    const float inv = 1.0f / rowsum;

    #pragma unroll
    for (int i = 0; i < 8; i++)
        prow[tid + i * 256] = (bf16_t)(e[i] * inv);
}

// ---------------------------------------------------------------------------
// Kernel 4: out = P @ V, B-fragments from transposed Vt[b][d][s].
// One wave = 32x64 tile, double-buffered like kernel 2 (K = 2048 keys).
// ---------------------------------------------------------------------------
__global__ __launch_bounds__(256) void pv_kernel(
    const bf16_t* __restrict__ p, const bf16_t* __restrict__ vt,
    float* __restrict__ out)
{
    const int lane = threadIdx.x & 31;
    const int wid  = blockIdx.x * 8 + (threadIdx.x >> 5); // 0 .. 4095
    const int b    = wid / (64 * 16);
    const int rem  = wid % (64 * 16);
    const int qt   = rem / 16;                            // 32-row query tile
    const int cg   = rem % 16;                            // 64-col group of D
    const int half = lane >> 4;

    const bf16_t* p0 = p + ((size_t)(b * SEQ + qt * 32 + (lane & 15))) * SEQ;
    const bf16_t* p1 = p0 + (size_t)16 * SEQ;
    const bf16_t* vr[4];
    #pragma unroll
    for (int t = 0; t < 4; t++) {
        const int col = cg * 64 + t * 16 + (lane & 15);
        vr[t] = vt + ((size_t)b * DIM + col) * SEQ;
    }

    v8f acc[2][4] = {};
    Stage s0, s1;
    load_stage(s0, p0, p1, vr[0], vr[1], vr[2], vr[3], half * 8, half * 16);
    SCHED_FENCE();
    for (int jt = 0; jt < SEQ; jt += 64) {
        load_stage(s1, p0, p1, vr[0], vr[1], vr[2], vr[3],
                   jt + 32 + half * 8, jt + 32 + half * 16);
        SCHED_FENCE();
        mma_stage(s0, acc);
        SCHED_FENCE();
        if (jt + 64 < SEQ)
            load_stage(s0, p0, p1, vr[0], vr[1], vr[2], vr[3],
                       jt + 64 + half * 8, jt + 64 + half * 16);
        SCHED_FENCE();
        mma_stage(s1, acc);
        SCHED_FENCE();
    }

    #pragma unroll
    for (int j = 0; j < 2; j++)
        #pragma unroll
        for (int t = 0; t < 4; t++) {
            const int col = cg * 64 + t * 16 + (lane & 15);
            #pragma unroll
            for (int r = 0; r < 8; r++) {
                const int qr = qt * 32 + j * 16 + half * 8 + r;
                out[((size_t)b * SEQ + qr) * DIM + col] = acc[j][t][r];
            }
        }
}

// ---------------------------------------------------------------------------
// Workspace layout (bytes):
//   [0,      16Mi)  Q   bf16 [B*S][D]
//   [16Mi,   32Mi)  K   bf16 [B*S][D]
//   [32Mi,   48Mi)  Vt  bf16 [B][D][S]
//   [48Mi,  112Mi)  scores f32 [B][S][S]
//   [112Mi, 144Mi)  P   bf16 [B][S][S]
// ---------------------------------------------------------------------------
extern "C" void kernel_launch(void* const* d_in, const int* in_sizes, int n_in,
                              void* d_out, int out_size, void* d_ws,
                              size_t ws_size, hipStream_t stream)
{
    (void)in_sizes; (void)n_in; (void)out_size; (void)ws_size;
    const float* X    = (const float*)d_in[0];
    const int*   mask = (const int*)d_in[1];
    const float* Wq   = (const float*)d_in[2];
    const float* bq   = (const float*)d_in[3];
    const float* Wk   = (const float*)d_in[4];
    const float* bk   = (const float*)d_in[5];
    const float* Wv   = (const float*)d_in[6];
    const float* bv   = (const float*)d_in[7];
    float* out = (float*)d_out;

    char* ws = (char*)d_ws;
    const size_t QB = (size_t)BATCH * SEQ * DIM * sizeof(bf16_t);   // 16 MiB
    const size_t SB = (size_t)BATCH * SEQ * SEQ * sizeof(float);    // 64 MiB
    bf16_t* qbuf  = (bf16_t*)(ws);
    bf16_t* kbuf  = (bf16_t*)(ws + QB);
    bf16_t* vtbuf = (bf16_t*)(ws + 2 * QB);
    float*  sc    = (float*)(ws + 3 * QB);
    bf16_t* pbuf  = (bf16_t*)(ws + 3 * QB + SB);

    qkv_proj_kernel<<<3072, 256, 0, stream>>>(X, Wq, bq, Wk, bk, Wv, bv,
                                              qbuf, kbuf, vtbuf);
    scores_kernel<<<1024, 256, 0, stream>>>(qbuf, kbuf, sc);
    softmax_kernel<<<BATCH * SEQ, 256, 0, stream>>>(sc, mask, pbuf);
    pv_kernel<<<512, 256, 0, stream>>>(pbuf, vtbuf, out);
}
